// SpatialLinearAttention_76029511074392
// MI455X (gfx1250) — matrix operand order repaired and checked by reading, not compiled, not run on hardware
//
#include <hip/hip_runtime.h>

// ---------------------------------------------------------------------------
// Types for CDNA5 WMMA (wave32)
// ---------------------------------------------------------------------------
typedef __bf16 bf16_t;
typedef bf16_t v16bf __attribute__((ext_vector_type(16)));
typedef bf16_t v8bf  __attribute__((ext_vector_type(8)));
typedef float  v8f   __attribute__((ext_vector_type(8)));

typedef int          i32x4 __attribute__((ext_vector_type(4)));
typedef int          i32x8 __attribute__((ext_vector_type(8)));
typedef unsigned int u32x4 __attribute__((ext_vector_type(4)));

#define DEV __device__ __forceinline__

// ---- gfx1250 data-movement feature detection (device pass only) ----
#if defined(__AMDGCN__) && defined(__gfx1250__)
  #if __has_builtin(__builtin_amdgcn_tensor_load_to_lds) && __has_builtin(__builtin_amdgcn_s_wait_tensorcnt)
    #define USE_TDM 1
  #endif
  #if __has_builtin(__builtin_amdgcn_global_load_async_to_lds_b128)
    #define USE_ASYNC 1
  #endif
#endif

#ifdef USE_TDM
// DMA a 2D bf16 tile (tileX elems per row, tileY rows, row stride in elems)
// from global into LDS at byte offset ldsOff, with LDS row padding:
// padInterval code: 3 -> pad after 16 DWORDs (64B), 4 -> after 32 DWORDs (128B)
// padAmount   code: 3 -> pad 4 DWORDs (16B)
DEV void tdm_load_2d_bf16(unsigned ldsOff, const unsigned short* gsrc,
                          unsigned tileX, unsigned tileY, unsigned strideElems,
                          unsigned padInterval, unsigned padAmount) {
  unsigned long long ga = (unsigned long long)gsrc;
  u32x4 g0;
  g0[0] = 1u;                                   // count=1, user descriptor
  g0[1] = ldsOff;                               // lds byte address
  g0[2] = (unsigned)(ga & 0xFFFFFFFFu);         // global_addr[31:0]
  g0[3] = (unsigned)((ga >> 32) & 0x1FFFFFFu)   // global_addr[56:32]
        | (2u << 30);                           // type=2 ("image")
  i32x8 g1;
  g1[0] = (int)((1u << 16)                      // data_size = 2 bytes
              | (1u << 20)                      // pad_enable
              | (padInterval << 22)
              | (padAmount << 25));
  g1[1] = (int)((tileX & 0xFFFFu) << 16);                         // tensor_dim0[15:0]
  g1[2] = (int)((tileX >> 16) | ((tileY & 0xFFFFu) << 16));       // dim0 hi | dim1 lo
  g1[3] = (int)((tileY >> 16) | ((tileX & 0xFFFFu) << 16));       // dim1 hi | tile_dim0
  g1[4] = (int)(tileY & 0xFFFFu);                                 // tile_dim1, tile_dim2=0
  g1[5] = (int)strideElems;                                       // tensor_dim0_stride lo
  g1[6] = 0;
  g1[7] = 0;
  i32x4 g2 = {0, 0, 0, 0};
  i32x4 g3 = {0, 0, 0, 0};
  i32x8 g4 = {0, 0, 0, 0, 0, 0, 0, 0};
  __builtin_amdgcn_tensor_load_to_lds(g0, g1, g2, g3, g4, 0);
}
DEV void tdm_wait0() { __builtin_amdgcn_s_wait_tensorcnt(0); }
DEV unsigned lds_off(const void* p) { return (unsigned)(unsigned long long)p; }
#endif

DEV void wait_async0() {
#if defined(__AMDGCN__) && defined(__gfx1250__) && __has_builtin(__builtin_amdgcn_s_wait_asynccnt)
  __builtin_amdgcn_s_wait_asynccnt(0);
#elif defined(USE_ASYNC)
  asm volatile("s_wait_asynccnt 0x0" ::: "memory");
#endif
}

// copy 16 bytes global -> LDS (async DMA if available, else through VGPRs)
DEV void cp16(unsigned short* l, const unsigned short* g) {
#ifdef USE_ASYNC
  __builtin_amdgcn_global_load_async_to_lds_b128(
      (__attribute__((address_space(1))) i32x4*)(unsigned long long)g,
      (__attribute__((address_space(3))) i32x4*)(unsigned)(unsigned long long)l,
      0, 0);
#else
  *(uint4*)l = *(const uint4*)g;
#endif
}

DEV unsigned short f2bf(float x) {
  union { float f; unsigned u; } v; v.f = x;
  unsigned r = v.u + 0x7FFFu + ((v.u >> 16) & 1u);   // round-to-nearest-even
  return (unsigned short)(r >> 16);
}
DEV float bf2f(unsigned short h) {
  union { unsigned u; float f; } v; v.u = ((unsigned)h) << 16;
  return v.f;
}
DEV v8f zero8() {
  v8f z = {0.f, 0.f, 0.f, 0.f, 0.f, 0.f, 0.f, 0.f};
  return z;
}

// A fragment: lane(m=lane&15, half=lane>>4) reads K = half*16..+15
DEV v16bf ld_frag_a(const unsigned short* p) {
  union { v16bf v; v8bf h[2]; } f;
  f.h[0] = *(const v8bf*)(p);
  f.h[1] = *(const v8bf*)(p + 8);
  return f.v;
}
// B fragment from [k_hi][n][k_lo] staging: 16 contiguous shorts = K half*16..+15
DEV v16bf ld_frag_b(const unsigned short* p) {
  union { v16bf v; v8bf h[2]; } f;
  f.h[0] = *(const v8bf*)(p);
  f.h[1] = *(const v8bf*)(p + 8);
  return f.v;
}
DEV v8f wmma_bf16(v16bf a, v16bf b, v8f c) {
  return __builtin_amdgcn_wmma_f32_16x16x32_bf16(
      false, a, false, b, (short)0, c, false, false);
}

DEV float waveMax(float x) {
  #pragma unroll
  for (int o = 16; o > 0; o >>= 1) x = fmaxf(x, __shfl_xor(x, o, 32));
  return x;
}
DEV float waveSum(float x) {
  #pragma unroll
  for (int o = 16; o > 0; o >>= 1) x += __shfl_xor(x, o, 32);
  return x;
}

// ---------------------------------------------------------------------------
// fp32 -> bf16 weight conversion
// ---------------------------------------------------------------------------
__global__ __launch_bounds__(256)
void cvt_f32_bf16_kernel(const float* __restrict__ s, unsigned short* __restrict__ d) {
  long i = ((long)blockIdx.x * 256 + threadIdx.x) * 4;
  float4 f = *(const float4*)(s + i);
  ushort4 u;
  u.x = f2bf(f.x); u.y = f2bf(f.y); u.z = f2bf(f.z); u.w = f2bf(f.w);
  *(ushort4*)(d + i) = u;
}

// ---------------------------------------------------------------------------
// Tiled GEMM: C[b] = A (bf16, MxK weights) * B[b] (KxN), N = 4096 fixed.
// Block tile 128x128, K step 32, 8 waves, wave tile 32x64 (2x4 WMMA tiles).
// ---------------------------------------------------------------------------
template<bool B_IS_F32, bool RESID>
__global__ __launch_bounds__(256)
void gemm128_kernel(const unsigned short* __restrict__ A,
                    const void* __restrict__ Bsrc,
                    void* __restrict__ Cdst,
                    const float* __restrict__ resid,
                    int K, long bStride, long cStride)
{
  const int N = 4096;
  __shared__ unsigned short As[128][40];      // [m][k]: 64B data + 16B pad per row
  __shared__ unsigned short Bs[2][128][24];   // [k_hi][n][k_lo]

  const int n0 = blockIdx.x * 128;
  const int m0 = blockIdx.y * 128;
  const int bz = blockIdx.z;

  const float*          Bf = (const float*)Bsrc + (long)bz * bStride;
  const unsigned short* Bh = (const unsigned short*)Bsrc + (long)bz * bStride;

  const int t     = threadIdx.x;
  const int lane  = t & 31;
  const int wv    = t >> 5;
  const int hf    = lane >> 4;
  const int ml    = lane & 15;
  const int mbase = (wv & 3) * 32;
  const int nbase = (wv >> 2) * 64;

  v8f acc[2][4];
  #pragma unroll
  for (int a = 0; a < 2; ++a)
    #pragma unroll
    for (int b = 0; b < 4; ++b) acc[a][b] = zero8();

  for (int k0 = 0; k0 < K; k0 += 32) {
    __syncthreads();
    // ---- stage A tile 128x32 ----
#ifdef USE_TDM
    if (wv == 0)
      tdm_load_2d_bf16(lds_off(&As[0][0]), A + (long)m0 * K + k0,
                       32, 128, (unsigned)K, 3, 3);
#else
    #pragma unroll
    for (int i = 0; i < 2; ++i) {
      int id = t + i * 256;
      int r  = id >> 2;
      int co = (id & 3) * 8;
      cp16(&As[r][co], A + (long)(m0 + r) * K + k0 + co);
    }
#endif
    // ---- stage B tile 32x128 into [k_hi][n][k_lo] (batched regs) ----
    ushort4 pk[4];
    #pragma unroll
    for (int i = 0; i < 4; ++i) {
      int id = t + i * 256;
      int n  = id & 127;
      int kb = (id >> 7) * 4;
      if (B_IS_F32) {
        const float* bp = Bf + (long)(k0 + kb) * N + (n0 + n);
        pk[i].x = f2bf(bp[0]);
        pk[i].y = f2bf(bp[N]);
        pk[i].z = f2bf(bp[2 * N]);
        pk[i].w = f2bf(bp[3 * N]);
        if (k0 + 32 < K) __builtin_prefetch(bp + (long)32 * N, 0, 1);
      } else {
        const unsigned short* bp = Bh + (long)(k0 + kb) * N + (n0 + n);
        pk[i].x = bp[0];
        pk[i].y = bp[N];
        pk[i].z = bp[2 * N];
        pk[i].w = bp[3 * N];
        if (k0 + 32 < K) __builtin_prefetch(bp + (long)32 * N, 0, 1);
      }
    }
    #pragma unroll
    for (int i = 0; i < 4; ++i) {
      int id = t + i * 256;
      int n  = id & 127;
      int kb = (id >> 7) * 4;
      *(ushort4*)&Bs[kb >> 4][n][kb & 15] = pk[i];
    }
#if defined(USE_TDM)
    if (wv == 0) tdm_wait0();
#elif defined(USE_ASYNC)
    wait_async0();
#endif
    __syncthreads();

    // ---- compute: 8 WMMA per wave per K-step ----
    v16bf afr[2];
    #pragma unroll
    for (int mt = 0; mt < 2; ++mt)
      afr[mt] = ld_frag_a(&As[mbase + mt * 16 + ml][hf * 16]);
    #pragma unroll
    for (int nt = 0; nt < 4; ++nt) {
      v16bf bfr = ld_frag_b(&Bs[hf][nbase + nt * 16 + ml][0]);
      #pragma unroll
      for (int mt = 0; mt < 2; ++mt)
        acc[mt][nt] = wmma_bf16(afr[mt], bfr, acc[mt][nt]);
    }
  }

  // ---- epilogue ----
  #pragma unroll
  for (int mt = 0; mt < 2; ++mt)
    #pragma unroll
    for (int nt = 0; nt < 4; ++nt)
      #pragma unroll
      for (int r = 0; r < 8; ++r) {
        int  mm  = m0 + mbase + mt * 16 + hf * 8 + r;
        int  nn  = n0 + nbase + nt * 16 + ml;
        long off = (long)mm * N + nn;
        if (RESID) {
          float* Cf = (float*)Cdst + (long)bz * cStride;
          Cf[off] = acc[mt][nt][r] + resid[(long)bz * cStride + off];
        } else {
          unsigned short* Ch = (unsigned short*)Cdst + (long)bz * cStride;
          Ch[off] = f2bf(acc[mt][nt][r]);
        }
      }
}

// ---------------------------------------------------------------------------
// Row softmax of k over N=4096, in place. One 256-thread block per row.
// ---------------------------------------------------------------------------
__global__ __launch_bounds__(256)
void softmax_k_kernel(unsigned short* __restrict__ qkv)
{
  __shared__ float red[8];
  int row = blockIdx.x;              // 0..8191
  int b   = row >> 9;
  int hh  = (row >> 6) & 7;
  int i   = row & 63;
  unsigned short* p = qkv + ((long)b * 1536 + hh * 192 + 64 + i) * 4096;
  int t = threadIdx.x;

  float vals[16];
  float mx = -3.402823e38f;
  #pragma unroll
  for (int j = 0; j < 16; ++j) {
    vals[j] = bf2f(p[t + j * 256]);
    mx = fmaxf(mx, vals[j]);
  }
  mx = waveMax(mx);
  if ((t & 31) == 0) red[t >> 5] = mx;
  __syncthreads();
  mx = red[0];
  #pragma unroll
  for (int w = 1; w < 8; ++w) mx = fmaxf(mx, red[w]);

  float s = 0.f;
  #pragma unroll
  for (int j = 0; j < 16; ++j) {
    vals[j] = __expf(vals[j] - mx);
    s += vals[j];
  }
  s = waveSum(s);
  __syncthreads();
  if ((t & 31) == 0) red[t >> 5] = s;
  __syncthreads();
  s = 0.f;
  #pragma unroll
  for (int w = 0; w < 8; ++w) s += red[w];
  float inv = 1.0f / s;
  #pragma unroll
  for (int j = 0; j < 16; ++j)
    p[t + j * 256] = f2bf(vals[j] * inv);
}

// ---------------------------------------------------------------------------
// Column softmax of q over d=64 (stride 4096), one thread per (b,h,n).
// ---------------------------------------------------------------------------
__global__ __launch_bounds__(256)
void softmax_q_kernel(unsigned short* __restrict__ qkv)
{
  long g  = (long)blockIdx.x * 256 + threadIdx.x;   // 0..524287
  int  n  = (int)(g & 4095);
  int  hh = (int)((g >> 12) & 7);
  int  b  = (int)(g >> 15);
  unsigned short* p = qkv + ((long)b * 1536 + hh * 192) * 4096 + n;

  float v[64];
  float mx = -3.402823e38f;
  #pragma unroll
  for (int i = 0; i < 64; ++i) {
    v[i] = bf2f(p[(long)i * 4096]);
    mx = fmaxf(mx, v[i]);
  }
  float s = 0.f;
  #pragma unroll
  for (int i = 0; i < 64; ++i) { v[i] = __expf(v[i] - mx); s += v[i]; }
  float inv = 1.0f / s;
  #pragma unroll
  for (int i = 0; i < 64; ++i) p[(long)i * 4096] = f2bf(v[i] * inv);
}

// ---------------------------------------------------------------------------
// ctx[i][j] = sum_n k_s[i,n] * v[j,n]  (per b,h).  Writes ctx^T (j,i) in bf16.
// ---------------------------------------------------------------------------
__global__ __launch_bounds__(256)
void ctx_kernel(const unsigned short* __restrict__ qkv,
                unsigned short* __restrict__ ctxT)
{
  __shared__ unsigned short Ks[64][40];      // [i][n_lo]: 64B data + 16B pad
  __shared__ unsigned short Vs[2][64][24];   // [n_hi][j][n_lo]

  int bh = blockIdx.x;                 // 0..127
  int b  = bh >> 3, h = bh & 7;
  const unsigned short* kb = qkv + ((long)b * 1536 + h * 192 +  64) * 4096;
  const unsigned short* vb = qkv + ((long)b * 1536 + h * 192 + 128) * 4096;

  const int t    = threadIdx.x;
  const int lane = t & 31;
  const int wv   = t >> 5;
  const int hf   = lane >> 4;
  const int ml   = lane & 15;
  const int mi   = wv & 3;             // i-tile
  const int jt0  = (wv >> 2) * 2;      // j-tiles jt0, jt0+1

  v8f acc[2] = {zero8(), zero8()};

  for (int n0 = 0; n0 < 4096; n0 += 32) {
    __syncthreads();
    // ---- stage k tile 64x32 ----
#ifdef USE_TDM
    if (wv == 0)
      tdm_load_2d_bf16(lds_off(&Ks[0][0]), kb + n0, 32, 64, 4096, 3, 3);
#else
    {
      int r  = t >> 2;
      int co = (t & 3) * 8;
      cp16(&Ks[r][co], kb + (long)r * 4096 + n0 + co);
    }
#endif
    // ---- stage v tile 64x32 -> transposed staging (batched regs) ----
    ushort4 vk[2];
    #pragma unroll
    for (int i = 0; i < 2; ++i) {
      int id = t + i * 256;
      int j  = id >> 3;
      int q  = (id & 7) * 4;
      vk[i] = *(const ushort4*)(vb + (long)j * 4096 + n0 + q);
    }
    #pragma unroll
    for (int i = 0; i < 2; ++i) {
      int id = t + i * 256;
      int j  = id >> 3;
      int q  = (id & 7) * 4;
      *(ushort4*)&Vs[q >> 4][j][q & 15] = vk[i];
    }
#if defined(USE_TDM)
    if (wv == 0) tdm_wait0();
#elif defined(USE_ASYNC)
    wait_async0();
#endif
    __syncthreads();

    v16bf a = ld_frag_a(&Ks[mi * 16 + ml][hf * 16]);
    #pragma unroll
    for (int tt = 0; tt < 2; ++tt) {
      v16bf bfr = ld_frag_b(&Vs[hf][(jt0 + tt) * 16 + ml][0]);
      acc[tt] = wmma_bf16(a, bfr, acc[tt]);
    }
  }

  #pragma unroll
  for (int tt = 0; tt < 2; ++tt)
    #pragma unroll
    for (int r = 0; r < 8; ++r) {
      int i = mi * 16 + hf * 8 + r;
      int j = (jt0 + tt) * 16 + ml;
      ctxT[(long)bh * 4096 + j * 64 + i] = f2bf(acc[tt][r]);
    }
}

// ---------------------------------------------------------------------------
// out[j][n] = sum_i ctxT[j][i] * q_s[i][n]   per (b,h): M=64, K=64, N tile 128.
// ---------------------------------------------------------------------------
__global__ __launch_bounds__(256)
void out_kernel(const unsigned short* __restrict__ qkv,
                const unsigned short* __restrict__ ctxT,
                unsigned short* __restrict__ outb)
{
  __shared__ unsigned short As[64][72];      // ctx^T [j][i]: 128B data + 16B pad
  __shared__ unsigned short Bs[4][128][24];  // q [i_hi][n][i_lo]

  int bh = blockIdx.y;
  int b  = bh >> 3, h = bh & 7;
  int n0 = blockIdx.x * 128;
  const unsigned short* qb = qkv + ((long)b * 1536 + h * 192) * 4096;
  const unsigned short* Ab = ctxT + (long)bh * 4096;

  const int t    = threadIdx.x;
  const int lane = t & 31;
  const int wv   = t >> 5;
  const int hf   = lane >> 4;
  const int ml   = lane & 15;
  const int mi   = wv & 3;             // j-tile
  const int ntb  = (wv >> 2) * 4;      // 4 n-tiles per wave

  // ---- stage A (full 64x64) ----
#ifdef USE_TDM
  if (wv == 0)
    tdm_load_2d_bf16(lds_off(&As[0][0]), Ab, 64, 64, 64, 4, 3);
#else
  #pragma unroll
  for (int i = 0; i < 2; ++i) {
    int id = t + i * 256;
    int r  = id >> 3;
    int co = (id & 7) * 8;
    cp16(&As[r][co], Ab + (long)r * 64 + co);
  }
#endif
  // ---- stage B (full 64x128) into [i_hi][n][i_lo] (batched regs) ----
  ushort4 pk[8];
  #pragma unroll
  for (int i = 0; i < 8; ++i) {
    int id = t + i * 256;
    int n  = id & 127;
    int kb = (id >> 7) * 4;
    const unsigned short* bp = qb + (long)kb * 4096 + n0 + n;
    pk[i].x = bp[0]; pk[i].y = bp[4096]; pk[i].z = bp[8192]; pk[i].w = bp[12288];
  }
  #pragma unroll
  for (int i = 0; i < 8; ++i) {
    int id = t + i * 256;
    int n  = id & 127;
    int kb = (id >> 7) * 4;
    *(ushort4*)&Bs[kb >> 4][n][kb & 15] = pk[i];
  }
#if defined(USE_TDM)
  if (wv == 0) tdm_wait0();
#elif defined(USE_ASYNC)
  wait_async0();
#endif
  __syncthreads();

  v8f acc[4] = {zero8(), zero8(), zero8(), zero8()};
  #pragma unroll
  for (int ks = 0; ks < 2; ++ks) {
    v16bf a = ld_frag_a(&As[mi * 16 + ml][ks * 32 + hf * 16]);
    #pragma unroll
    for (int tt = 0; tt < 4; ++tt) {
      v16bf bfr = ld_frag_b(&Bs[ks * 2 + hf][(ntb + tt) * 16 + ml][0]);
      acc[tt] = wmma_bf16(a, bfr, acc[tt]);
    }
  }

  #pragma unroll
  for (int tt = 0; tt < 4; ++tt)
    #pragma unroll
    for (int r = 0; r < 8; ++r) {
      int j = mi * 16 + hf * 8 + r;
      int n = n0 + (ntb + tt) * 16 + ml;
      outb[((long)b * 512 + h * 64 + j) * 4096 + n] = f2bf(acc[tt][r]);
    }
}

// ---------------------------------------------------------------------------
// Launcher
// ---------------------------------------------------------------------------
extern "C" void kernel_launch(void* const* d_in, const int* in_sizes, int n_in,
                              void* d_out, int out_size, void* d_ws, size_t ws_size,
                              hipStream_t stream) {
  (void)in_sizes; (void)n_in; (void)out_size; (void)ws_size;
  const float* x     = (const float*)d_in[0];   // [16][512][64][64]
  const float* wqkv  = (const float*)d_in[1];   // [1536][512]
  const float* wproj = (const float*)d_in[2];   // [512][512]
  float*       out   = (float*)d_out;

  unsigned short* ws      = (unsigned short*)d_ws;
  unsigned short* wqkv_h  = ws;
  unsigned short* wproj_h = wqkv_h + (long)1536 * 512;
  unsigned short* qkv_h   = wproj_h + (long)512 * 512;
  unsigned short* ctxT_h  = qkv_h + (long)16 * 1536 * 4096;
  unsigned short* outb_h  = ctxT_h + (long)128 * 64 * 64;

  cvt_f32_bf16_kernel<<<768, 256, 0, stream>>>(wqkv, wqkv_h);
  cvt_f32_bf16_kernel<<<256, 256, 0, stream>>>(wproj, wproj_h);

  gemm128_kernel<true, false><<<dim3(32, 12, 16), 256, 0, stream>>>(
      wqkv_h, x, qkv_h, nullptr, 512, (long)512 * 4096, (long)1536 * 4096);

  softmax_k_kernel<<<8192, 256, 0, stream>>>(qkv_h);
  softmax_q_kernel<<<2048, 256, 0, stream>>>(qkv_h);

  ctx_kernel<<<128, 256, 0, stream>>>(qkv_h, ctxT_h);
  out_kernel<<<dim3(32, 128), 256, 0, stream>>>(qkv_h, ctxT_h, outb_h);

  gemm128_kernel<false, true><<<dim3(32, 4, 16), 256, 0, stream>>>(
      wproj_h, outb_h, out, x, 512, (long)512 * 4096, (long)512 * 4096);
}
